// CMPGNN_Net_50190987821197
// MI455X (gfx1250) — compile-verified, matrix operand order checked by reading, not run on hardware
//
#include <hip/hip_runtime.h>
#include <math.h>

// Problem constants (match reference)
#define NN   10000
#define EE   100000
#define GG   64
#define FF   64
#define DD   256
#define LL   3
#define HIDD 256
#define NCLS 10
#define EPS  1e-5f
#define SLOPE 0.01f

typedef __attribute__((ext_vector_type(16))) _Float16 v16h;
typedef __attribute__((ext_vector_type(8)))  float    v8f;

__device__ __forceinline__ float sigm(float v) { return 1.0f / (1.0f + __expf(-v)); }
__device__ __forceinline__ float lrelu(float v) { return v > 0.0f ? v : SLOPE * v; }

// ---------------------------------------------------------------------------
// Weight prep: f32 [K,N] row-major  ->  f16 [N,K] row-major (so B fragments are
// contiguous 8-half chunks along K).
// ---------------------------------------------------------------------------
__global__ void transpose_to_half(const float* __restrict__ W, _Float16* __restrict__ Wt,
                                  int K, int N) {
  int idx = blockIdx.x * 256 + threadIdx.x;
  if (idx < K * N) {
    int k = idx / N, n = idx - k * N;
    Wt[(size_t)n * K + k] = (_Float16)W[idx];
  }
}

__global__ void fill_kernel(float* __restrict__ p, float v, int n) {
  int i = blockIdx.x * 256 + threadIdx.x;
  if (i < n) p[i] = v;
}

// ---------------------------------------------------------------------------
// WMMA GEMM:  out[M,Nout] = epi( A[M,K] @ W[K,Nout] + bias )
// One wave (32 lanes) computes a 16 x (16*NT) strip using v_wmma_f32_16x16x32_f16.
// A-tile layout (16-bit A, 16x32): lane l -> row m = l&15, k-chunk base (l>>4)*8;
// halfs[0..7] = A[m][k+kb .. +7], halfs[8..15] = A[m][k+16+kb .. +7].
// B mirrors it on Wt[N][K]. C/D: lane l col n=l&15; VGPR r -> row r + 8*(l>>4).
//
// AMODE 0: A0 is plain [M,K] of type TA (float or _Float16)
// AMODE 1: row m = concat( x[m] (256), gf[batch[m]] (256) ), K=512 (TA=float)
// AMODE 2: row e = concat( x[dst[e]], x[src[e]], gf[batch[dst[e]]] ), K=768 (TA=float)
// EPI 0: identity   1: sigmoid   2: leaky-relu   3: sigmoid(v)*gmul[m,n]
// Output written as TO (float or _Float16).
// ---------------------------------------------------------------------------
template <int AMODE, int EPI, int NT, typename TA, typename TO>
__global__ __launch_bounds__(32) void wmma_gemm(
    const TA* __restrict__ A0, const TA* __restrict__ A1,
    const int* __restrict__ dstIdx, const int* __restrict__ srcIdx,
    const int* __restrict__ batchIdx,
    const _Float16* __restrict__ Wt, const float* __restrict__ bias,
    const TA* __restrict__ gmul, TO* __restrict__ out,
    int M, int K, int Nout) {
  const int lane   = threadIdx.x & 31;
  const int mtile  = blockIdx.x * 16;
  const int ntile0 = blockIdx.y * (16 * NT);
  const int mrow   = mtile + (lane & 15);
  const int kb     = (lane >> 4) * 8;

  const TA* base0 = nullptr;
  const TA* base1 = nullptr;
  const TA* base2 = nullptr;
  if constexpr (AMODE == 0) {
    base0 = A0 + (size_t)mrow * K;
  } else if constexpr (AMODE == 1) {
    base0 = A0 + (size_t)mrow * DD;
    base1 = A1 + (size_t)batchIdx[mrow] * DD;
  } else {
    int di = dstIdx[mrow];
    base0 = A0 + (size_t)di * DD;
    base1 = A0 + (size_t)srcIdx[mrow] * DD;
    base2 = A1 + (size_t)batchIdx[di] * DD;
    // warm the randomly gathered rows (lowers to global_prefetch_b8)
    __builtin_prefetch(base0, 0, 1);
    __builtin_prefetch(base1, 0, 1);
    __builtin_prefetch(base2, 0, 1);
  }

  v8f acc[NT];
#pragma unroll
  for (int t = 0; t < NT; ++t)
#pragma unroll
    for (int j = 0; j < 8; ++j) acc[t][j] = 0.0f;

  for (int k = 0; k < K; k += 32) {
    v16h a;
#pragma unroll
    for (int hf = 0; hf < 2; ++hf) {
      const int c = k + kb + hf * 16;  // 8-aligned; never straddles a 256 boundary
      const TA* p;
      if constexpr (AMODE == 0) {
        p = base0 + c;
      } else if constexpr (AMODE == 1) {
        p = (c < DD) ? (base0 + c) : (base1 + (c - DD));
      } else {
        const int seg = c >> 8;
        p = (seg == 0 ? base0 : (seg == 1 ? base1 : base2)) + (c & 255);
      }
#pragma unroll
      for (int j = 0; j < 8; ++j) a[hf * 8 + j] = (_Float16)p[j];
    }
#pragma unroll
    for (int t = 0; t < NT; ++t) {
      const int ncol = ntile0 + t * 16 + (lane & 15);
      const _Float16* w0 = Wt + (size_t)ncol * K + k + kb;
      v16h bfrag;
#pragma unroll
      for (int j = 0; j < 8; ++j) { bfrag[j] = w0[j]; bfrag[8 + j] = w0[16 + j]; }
      acc[t] = __builtin_amdgcn_wmma_f32_16x16x32_f16(
          false, a, false, bfrag, (short)0, acc[t], false, false);
    }
  }

#pragma unroll
  for (int t = 0; t < NT; ++t) {
    const int ncol = ntile0 + t * 16 + (lane & 15);
    const float bv = bias[ncol];
#pragma unroll
    for (int r = 0; r < 8; ++r) {
      const int mr = mtile + r + 8 * (lane >> 4);
      float v = acc[t][r] + bv;
      if constexpr (EPI == 1) v = sigm(v);
      else if constexpr (EPI == 2) v = lrelu(v);
      else if constexpr (EPI == 3) v = sigm(v) * (float)gmul[(size_t)mr * Nout + ncol];
      out[(size_t)mr * Nout + ncol] = (TO)v;
    }
  }
}

// ---------------------------------------------------------------------------
// Readout pieces
// ---------------------------------------------------------------------------
__global__ __launch_bounds__(256) void readout_reduce(
    const _Float16* __restrict__ gw, const float* __restrict__ x,
    const int* __restrict__ batch, float* __restrict__ gfnew,
    float* __restrict__ gsum, float* __restrict__ cnt) {
  const int n = blockIdx.x, d = threadIdx.x;
  const size_t base = (size_t)n * DD;
  const float w = (float)gw[base + d];
  const int g = batch[n];
  atomicAdd(&gfnew[(size_t)g * DD + d], w * x[base + d]);
  __shared__ float red[DD];
  red[d] = w * w;
  __syncthreads();
  for (int s = DD / 2; s > 0; s >>= 1) {
    if (d < s) red[d] += red[d + s];
    __syncthreads();
  }
  if (d == 0) {
    atomicAdd(&gsum[g], sqrtf(red[0]));
    atomicAdd(&cnt[g], 1.0f);
  }
}

__global__ __launch_bounds__(256) void gru_kernel(
    const float* __restrict__ xin, const float* __restrict__ h,
    const float* __restrict__ wih, const float* __restrict__ whh,
    const float* __restrict__ bih, const float* __restrict__ bhh,
    float* __restrict__ hout) {
  const int g = blockIdx.x, d = threadIdx.x;
  __shared__ float xs[DD], hs[DD];
  xs[d] = xin[(size_t)g * DD + d];
  hs[d] = h[(size_t)g * DD + d];
  __syncthreads();
  float ir = bih[d], iz = bih[DD + d], inn = bih[2 * DD + d];
  float hr = bhh[d], hz = bhh[DD + d], hn = bhh[2 * DD + d];
  const float* wi0 = wih + (size_t)d * DD;
  const float* wi1 = wih + (size_t)(DD + d) * DD;
  const float* wi2 = wih + (size_t)(2 * DD + d) * DD;
  const float* wh0 = whh + (size_t)d * DD;
  const float* wh1 = whh + (size_t)(DD + d) * DD;
  const float* wh2 = whh + (size_t)(2 * DD + d) * DD;
  for (int k = 0; k < DD; ++k) {
    const float xv = xs[k], hv = hs[k];
    ir += xv * wi0[k]; iz += xv * wi1[k]; inn += xv * wi2[k];
    hr += hv * wh0[k]; hz += hv * wh1[k]; hn += hv * wh2[k];
  }
  const float r = sigm(ir + hr);
  const float z = sigm(iz + hz);
  const float nv = tanhf(inn + r * hn);
  hout[(size_t)g * DD + d] = (1.0f - z) * nv + z * hs[d];
}

__global__ __launch_bounds__(GG) void ggl_kernel(const float* __restrict__ gsum,
                                                 const float* __restrict__ cnt,
                                                 float* __restrict__ out) {
  __shared__ float red[GG];
  const int g = threadIdx.x;
  red[g] = gsum[g] / fmaxf(cnt[g], 1.0f);
  __syncthreads();
  for (int s = GG / 2; s > 0; s >>= 1) {
    if (g < s) red[g] += red[g + s];
    __syncthreads();
  }
  if (g == 0) out[0] = red[0] / (float)GG;
}

// ---------------------------------------------------------------------------
// Edge aggregation: x_next += scatter_add(lw*msg, dst);  lgl[e,col] = ||lw[e]||
// ---------------------------------------------------------------------------
__global__ __launch_bounds__(256) void edge_scatter(
    const _Float16* __restrict__ lw, const _Float16* __restrict__ msg,
    const int* __restrict__ dst, float* __restrict__ xnext,
    float* __restrict__ lgl, int col) {
  const int e = blockIdx.x, d = threadIdx.x;
  const size_t base = (size_t)e * DD;
  const float l = (float)lw[base + d];
  const float m = (float)msg[base + d];
  atomicAdd(&xnext[(size_t)dst[e] * DD + d], l * m);
  __shared__ float red[DD];
  red[d] = l * l;
  __syncthreads();
  for (int s = DD / 2; s > 0; s >>= 1) {
    if (d < s) red[d] += red[d + s];
    __syncthreads();
  }
  if (d == 0) lgl[(size_t)e * LL + col] = sqrtf(red[0]);
}

// ---------------------------------------------------------------------------
// Classifier
// ---------------------------------------------------------------------------
__global__ __launch_bounds__(DD) void bn_stats(const float* __restrict__ gf,
                                               float* __restrict__ mu,
                                               float* __restrict__ var) {
  const int d = threadIdx.x;
  float s = 0.0f, s2 = 0.0f;
  for (int g = 0; g < GG; ++g) {
    const float v = gf[(size_t)g * DD + d];
    s += v; s2 += v * v;
  }
  const float m = s / (float)GG;
  mu[d] = m;
  var[d] = s2 / (float)GG - m * m;
}

__global__ __launch_bounds__(HIDD) void clf1_kernel(
    const float* __restrict__ gf, const float* __restrict__ mu,
    const float* __restrict__ var, const float* __restrict__ bng,
    const float* __restrict__ bnb, const float* __restrict__ W,
    const float* __restrict__ b, float* __restrict__ h) {
  const int g = blockIdx.x, j = threadIdx.x;
  __shared__ float xn[DD];
  xn[j] = (gf[(size_t)g * DD + j] - mu[j]) * rsqrtf(var[j] + EPS) * bng[j] + bnb[j];
  __syncthreads();
  float s = b[j];
  for (int k = 0; k < DD; ++k) s += xn[k] * W[(size_t)k * HIDD + j];
  h[(size_t)g * HIDD + j] = lrelu(s);
}

__global__ __launch_bounds__(32) void clf2_kernel(const float* __restrict__ h,
                                                  const float* __restrict__ W,
                                                  const float* __restrict__ b,
                                                  float* __restrict__ out) {
  const int g = blockIdx.x, c = threadIdx.x;
  __shared__ float lg[NCLS];
  if (c < NCLS) {
    float s = b[c];
    for (int k = 0; k < HIDD; ++k) s += h[(size_t)g * HIDD + k] * W[(size_t)k * NCLS + c];
    lg[c] = s;
  }
  __syncthreads();
  if (c < NCLS) {
    float mx = lg[0];
    for (int j = 1; j < NCLS; ++j) mx = fmaxf(mx, lg[j]);
    float se = 0.0f;
    for (int j = 0; j < NCLS; ++j) se += __expf(lg[j] - mx);
    out[(size_t)g * NCLS + c] = lg[c] - mx - __logf(se);
  }
}

// ---------------------------------------------------------------------------
// Host driver
// ---------------------------------------------------------------------------
extern "C" void kernel_launch(void* const* d_in, const int* in_sizes, int n_in,
                              void* d_out, int out_size, void* d_ws, size_t ws_size,
                              hipStream_t stream) {
  (void)in_sizes; (void)n_in; (void)out_size; (void)ws_size;
  const float* x_in    = (const float*)d_in[0];
  const int*   ei      = (const int*)d_in[1];
  const int*   batch   = (const int*)d_in[2];
  const float* fc_w    = (const float*)d_in[3];
  const float* fc_b    = (const float*)d_in[4];
  const float* mg1_w   = (const float*)d_in[5];
  const float* mg1_b   = (const float*)d_in[6];
  const float* mg2_w   = (const float*)d_in[7];
  const float* mg2_b   = (const float*)d_in[8];
  const float* gate_w  = (const float*)d_in[9];
  const float* gate_b  = (const float*)d_in[10];
  const float* rd_w    = (const float*)d_in[11];
  const float* rd_b    = (const float*)d_in[12];
  const float* gru_wih = (const float*)d_in[13];
  const float* gru_whh = (const float*)d_in[14];
  const float* gru_bih = (const float*)d_in[15];
  const float* gru_bhh = (const float*)d_in[16];
  const float* bn_g    = (const float*)d_in[17];
  const float* bn_b    = (const float*)d_in[18];
  const float* clf1_w  = (const float*)d_in[19];
  const float* clf1_b  = (const float*)d_in[20];
  const float* clf2_w  = (const float*)d_in[21];
  const float* clf2_b  = (const float*)d_in[22];

  const int* src = ei;        // edge_index[0]
  const int* dst = ei + EE;   // edge_index[1]

  // ---- workspace carve (256B aligned) ----
  char* ws = (char*)d_ws;
  size_t off = 0;
  auto carve = [&](size_t bytes) -> char* {
    off = (off + 255) & ~(size_t)255;
    char* p = ws + off;
    off += bytes;
    return p;
  };
  _Float16* fcWt   = (_Float16*)carve((size_t)DD * FF * 2);
  _Float16* rdWt   = (_Float16*)carve((size_t)DD * 2 * DD * 2);
  _Float16* mg1Wt  = (_Float16*)carve((size_t)LL * DD * 3 * DD * 2);
  _Float16* mg2Wt  = (_Float16*)carve((size_t)LL * DD * DD * 2);
  _Float16* gateWt = (_Float16*)carve((size_t)LL * DD * DD * 2);
  float* xcur  = (float*)carve((size_t)NN * DD * 4);
  float* xnext = (float*)carve((size_t)NN * DD * 4);
  float* gfA   = (float*)carve((size_t)GG * DD * 4);
  float* gfB   = (float*)carve((size_t)GG * DD * 4);
  float* gfnew = (float*)carve(((size_t)GG * DD + 2 * GG) * 4);
  float* gsum  = gfnew + GG * DD;
  float* cnt   = gsum + GG;
  float* mu    = (float*)carve((size_t)2 * DD * 4);
  float* var   = mu + DD;
  float* hbuf  = (float*)carve((size_t)GG * HIDD * 4);
  _Float16* h1  = (_Float16*)carve((size_t)EE * DD * 2);  // reused as gw and lw
  _Float16* msg = (_Float16*)carve((size_t)EE * DD * 2);

  float* out        = (float*)d_out;
  float* out_logits = out;                                // [G, NCLS]
  float* out_lgl    = out + GG * NCLS;                    // [E, L]
  float* out_ggl    = out + GG * NCLS + (size_t)EE * LL;  // [L+1]

  // ---- weight prep ----
  transpose_to_half<<<(FF * DD + 255) / 256, 256, 0, stream>>>(fc_w, fcWt, FF, DD);
  transpose_to_half<<<(2 * DD * DD + 255) / 256, 256, 0, stream>>>(rd_w, rdWt, 2 * DD, DD);
  for (int l = 0; l < LL; ++l) {
    transpose_to_half<<<(3 * DD * DD + 255) / 256, 256, 0, stream>>>(
        mg1_w + (size_t)l * 3 * DD * DD, mg1Wt + (size_t)l * 3 * DD * DD, 3 * DD, DD);
    transpose_to_half<<<(DD * DD + 255) / 256, 256, 0, stream>>>(
        mg2_w + (size_t)l * DD * DD, mg2Wt + (size_t)l * DD * DD, DD, DD);
    transpose_to_half<<<(DD * DD + 255) / 256, 256, 0, stream>>>(
        gate_w + (size_t)l * DD * DD, gateWt + (size_t)l * DD * DD, DD, DD);
  }
  fill_kernel<<<(GG * DD + 255) / 256, 256, 0, stream>>>(gfA, 0.0f, GG * DD);

  // ---- input projection: xcur = x @ fc_w + fc_b ----
  wmma_gemm<0, 0, 4, float, float><<<dim3(NN / 16, DD / 64), 32, 0, stream>>>(
      x_in, nullptr, nullptr, nullptr, nullptr, fcWt, fc_b, nullptr, xcur, NN, FF, DD);

  float* gf_cur = gfA; float* gf_nxt = gfB;
  float* x_cur = xcur; float* x_nxt = xnext;

  for (int i = 0; i <= LL; ++i) {
    // ---- readout ----
    wmma_gemm<1, 1, 4, float, _Float16><<<dim3(NN / 16, DD / 64), 32, 0, stream>>>(
        x_cur, gf_cur, nullptr, nullptr, batch, rdWt, rd_b, nullptr,
        h1 /*gw*/, NN, 2 * DD, DD);
    fill_kernel<<<(GG * DD + 2 * GG + 255) / 256, 256, 0, stream>>>(
        gfnew, 0.0f, GG * DD + 2 * GG);
    readout_reduce<<<NN, DD, 0, stream>>>(h1, x_cur, batch, gfnew, gsum, cnt);
    gru_kernel<<<GG, DD, 0, stream>>>(gfnew, gf_cur, gru_wih, gru_whh,
                                      gru_bih, gru_bhh, gf_nxt);
    ggl_kernel<<<1, GG, 0, stream>>>(gsum, cnt, out_ggl + i);
    { float* t = gf_cur; gf_cur = gf_nxt; gf_nxt = t; }

    if (i == LL) break;

    // ---- message block i ----
    wmma_gemm<2, 2, 4, float, _Float16><<<dim3(EE / 16, DD / 64), 32, 0, stream>>>(
        x_cur, gf_cur, dst, src, batch,
        mg1Wt + (size_t)i * 3 * DD * DD, mg1_b + (size_t)i * DD, nullptr,
        h1, EE, 3 * DD, DD);
    wmma_gemm<0, 2, 4, _Float16, _Float16><<<dim3(EE / 16, DD / 64), 32, 0, stream>>>(
        h1, nullptr, nullptr, nullptr, nullptr,
        mg2Wt + (size_t)i * DD * DD, mg2_b + (size_t)i * DD, nullptr,
        msg, EE, DD, DD);
    wmma_gemm<0, 3, 4, _Float16, _Float16><<<dim3(EE / 16, DD / 64), 32, 0, stream>>>(
        msg, nullptr, nullptr, nullptr, nullptr,
        gateWt + (size_t)i * DD * DD, gate_b + (size_t)i * DD, msg,
        h1 /*lw*/, EE, DD, DD);
    fill_kernel<<<(NN * DD + 255) / 256, 256, 0, stream>>>(x_nxt, 0.0f, NN * DD);
    edge_scatter<<<EE, DD, 0, stream>>>(h1, msg, dst, x_nxt, out_lgl, i);
    { float* t = x_cur; x_cur = x_nxt; x_nxt = t; }
  }

  // ---- classifier ----
  bn_stats<<<1, DD, 0, stream>>>(gf_cur, mu, var);
  clf1_kernel<<<GG, HIDD, 0, stream>>>(gf_cur, mu, var, bn_g, bn_b,
                                       clf1_w, clf1_b, hbuf);
  clf2_kernel<<<GG, 32, 0, stream>>>(hbuf, clf2_w, clf2_b, out_logits);
}
